// MultiHeadLatentAttention_67662914781406
// MI455X (gfx1250) — matrix operand order repaired and checked by reading, not compiled, not run on hardware
//
#include <hip/hip_runtime.h>
#include <math.h>

typedef __bf16 bf16;
typedef __attribute__((ext_vector_type(16))) __bf16 v16bf;
typedef __attribute__((ext_vector_type(8)))  __bf16 v8bf;
typedef __attribute__((ext_vector_type(8)))  float  v8f;

#define B_   2
#define T_   2048
#define D_   2048
#define LAT_ 512
#define PAST_ 2048
#define S_   4096
#define H_   16
#define HD_  128
#define PASTLEN_ 2048

#define DEV __device__ __forceinline__

DEV v8f vzero8() {
  v8f z;
  #pragma unroll
  for (int i = 0; i < 8; ++i) z[i] = 0.0f;
  return z;
}

// A-operand fragment: 16x32 bf16, row-major source (row r = lane&15).
// Lane half selects K offset 0/8 within each 16-wide K group (ISA 16-bit A table).
DEV v16bf load_frag_a(const bf16* __restrict__ tile, int ld) {
  const int lane = threadIdx.x & 31;
  const int r    = lane & 15;
  const int koff = (lane >> 4) << 3;          // 0 or 8
  const bf16* p = tile + (size_t)r * ld + koff;
  v8bf lo = *reinterpret_cast<const v8bf*>(p);        // K = koff .. koff+7
  v8bf hi = *reinterpret_cast<const v8bf*>(p + 16);   // K = 16+koff .. 16+koff+7
  v16bf f;
  #pragma unroll
  for (int i = 0; i < 8; ++i) { f[i] = lo[i]; f[i + 8] = hi[i]; }
  return f;
}

// B-operand fragment: 32x16 bf16 taken from Bt (N x K row-major).
// Lane holds column n = lane&15; lane half selects K offset 0/8 within each 16-wide K group.
DEV v16bf load_frag_b(const bf16* __restrict__ bt, int ld) {
  const int lane = threadIdx.x & 31;
  const int n  = lane & 15;
  const int kb = (lane >> 4) << 3;            // 0 or 8
  const bf16* p = bt + (size_t)n * ld + kb;
  v8bf lo = *reinterpret_cast<const v8bf*>(p);
  v8bf hi = *reinterpret_cast<const v8bf*>(p + 16);
  v16bf f;
  #pragma unroll
  for (int i = 0; i < 8; ++i) { f[i] = lo[i]; f[i + 8] = hi[i]; }
  return f;
}

DEV v8f wmma_bf16(v16bf a, v16bf b, v8f c) {
  return __builtin_amdgcn_wmma_f32_16x16x32_bf16(false, a, false, b, (short)0, c,
                                                 false, false);
}

// ---------------- elementwise helpers ----------------

__global__ __launch_bounds__(256) void convert_f32_bf16(const float* __restrict__ in,
                                                        bf16* __restrict__ out, size_t n) {
  size_t i = (size_t)blockIdx.x * 256 + threadIdx.x;
  if (i < n) out[i] = (bf16)in[i];
}

// in: R x C f32 (row-major); out: C x R bf16 (row-major) == in^T
__global__ __launch_bounds__(256) void transpose_f32_bf16(const float* __restrict__ in,
                                                          bf16* __restrict__ out,
                                                          int R, int C) {
  size_t i = (size_t)blockIdx.x * 256 + threadIdx.x;
  if (i >= (size_t)R * C) return;
  int r = (int)(i / C), c = (int)(i % C);
  out[(size_t)c * R + r] = (bf16)in[i];
}

__global__ __launch_bounds__(256) void copy_prev_latent(const float* __restrict__ prev,
                                                        bf16* __restrict__ latent) {
  size_t i = (size_t)blockIdx.x * 256 + threadIdx.x;
  const size_t per_b = (size_t)PAST_ * LAT_;
  if (i >= (size_t)B_ * per_b) return;
  size_t b = i / per_b, rem = i % per_b;
  latent[b * (size_t)S_ * LAT_ + rem] = (bf16)prev[i];
}

// one block per (b*t) row; 256 threads handle 512 elements
__global__ __launch_bounds__(256) void layernorm_latent(const float* __restrict__ xin,
                                                        const float* __restrict__ g,
                                                        const float* __restrict__ bta,
                                                        bf16* __restrict__ latent) {
  const int row = blockIdx.x;                // 0 .. B*T-1
  const int b = row / T_, tt = row % T_;
  const float* xr = xin + (size_t)row * LAT_;
  const int tid = threadIdx.x;
  __shared__ float s1[256], s2[256];
  float v0 = xr[tid], v1 = xr[tid + 256];
  s1[tid] = v0 + v1;
  s2[tid] = v0 * v0 + v1 * v1;
  __syncthreads();
  for (int o = 128; o > 0; o >>= 1) {
    if (tid < o) { s1[tid] += s1[tid + o]; s2[tid] += s2[tid + o]; }
    __syncthreads();
  }
  const float mu   = s1[0] * (1.0f / LAT_);
  const float var  = s2[0] * (1.0f / LAT_) - mu * mu;
  const float rstd = rsqrtf(var + 1e-5f);
  bf16* out = latent + ((size_t)b * S_ + PAST_ + tt) * LAT_;
  out[tid]       = (bf16)((v0 - mu) * rstd * g[tid]       + bta[tid]);
  out[tid + 256] = (bf16)((v1 - mu) * rstd * g[tid + 256] + bta[tid + 256]);
}

// ---------------- WMMA GEMM ----------------
// C[M x N] = A[M x K] * Bt[N x K]^T
// 4 waves (128 threads); 64x64 per wave (4x4 accumulators -> 32 FLOP/byte);
// block tile 128x128.
template <bool OUT_BF16, bool TRANS>
__global__ __launch_bounds__(128) void wmma_gemm(const bf16* __restrict__ A,
                                                 const bf16* __restrict__ Bt,
                                                 void* __restrict__ Cout,
                                                 int M, int N, int K,
                                                 int lda, int ldb, int ldc) {
  const int wave = threadIdx.x >> 5;
  const int lane = threadIdx.x & 31;
  const int m0 = blockIdx.x * 128 + (wave >> 1) * 64;
  const int n0 = blockIdx.y * 128 + (wave & 1) * 64;
  if (m0 >= M || n0 >= N) return;

  v8f acc[4][4];
  #pragma unroll
  for (int i = 0; i < 4; ++i)
    #pragma unroll
    for (int j = 0; j < 4; ++j) acc[i][j] = vzero8();

  for (int k = 0; k < K; k += 32) {
    if (k + 32 < K) {
      __builtin_prefetch(A + (size_t)m0 * lda + k + 32, 0, 0);
      __builtin_prefetch(Bt + (size_t)n0 * ldb + k + 32, 0, 0);
    }
    v16bf a[4], b[4];
    #pragma unroll
    for (int i = 0; i < 4; ++i)
      a[i] = load_frag_a(A + (size_t)(m0 + 16 * i) * lda + k, lda);
    #pragma unroll
    for (int j = 0; j < 4; ++j)
      b[j] = load_frag_b(Bt + (size_t)(n0 + 16 * j) * ldb + k, ldb);
    #pragma unroll
    for (int i = 0; i < 4; ++i)
      #pragma unroll
      for (int j = 0; j < 4; ++j)
        acc[i][j] = wmma_bf16(a[i], b[j], acc[i][j]);
  }

  const int rh = (lane >> 4) << 3;  // row offset 0/8 from lane half
  #pragma unroll
  for (int i = 0; i < 4; ++i)
    #pragma unroll
    for (int j = 0; j < 4; ++j)
      #pragma unroll
      for (int v = 0; v < 8; ++v) {
        const int m = m0 + i * 16 + v + rh;
        const int n = n0 + j * 16 + (lane & 15);
        const float val = acc[i][j][v];
        if (TRANS)         ((bf16*)Cout)[(size_t)n * ldc + m] = (bf16)val;
        else if (OUT_BF16) ((bf16*)Cout)[(size_t)m * ldc + n] = (bf16)val;
        else               ((float*)Cout)[(size_t)m * ldc + n] = val;
      }
}

// ---------------- Flash-style MLA attention ----------------
// grid (T/64, H, B), 128 threads = 4 waves, each wave owns 16 query rows.
// q: (B,T,D) bf16 ; k: (B,S,D) bf16 ; vT: (B,D,S) bf16 ; out: (B,T,D) bf16
__global__ __launch_bounds__(128) void mla_attention(const bf16* __restrict__ q,
                                                     const bf16* __restrict__ k,
                                                     const bf16* __restrict__ vT,
                                                     bf16* __restrict__ out) {
  const int lane = threadIdx.x & 31;
  const int wave = threadIdx.x >> 5;           // 0..3
  const int bi = blockIdx.z, h = blockIdx.y;
  const int qr0 = blockIdx.x * 64 + wave * 16;

  __shared__ bf16 pbuf[4][16][40];             // per-wave 16x32 P tile (padded)

  const bf16* qbase = q + ((size_t)(bi * T_ + qr0)) * D_ + h * HD_;
  const bf16* kbase = k + (size_t)bi * S_ * D_ + h * HD_;
  const bf16* vbase = vT + (size_t)bi * D_ * S_ + (size_t)(h * HD_) * S_;

  v16bf aq[4];
  #pragma unroll
  for (int kk = 0; kk < 4; ++kk) aq[kk] = load_frag_a(qbase + kk * 32, D_);

  v8f oacc[8];
  #pragma unroll
  for (int cb = 0; cb < 8; ++cb) oacc[cb] = vzero8();
  float rmax[8], rsum[8];
  #pragma unroll
  for (int v = 0; v < 8; ++v) { rmax[v] = -1e30f; rsum[v] = 0.0f; }

  const float scale = 0.08838834764831845f;    // 1/sqrt(128)
  const int rh = (lane >> 4) << 3;
  const int jmax = PASTLEN_ + qr0 + 15;

  for (int s0 = 0; s0 < S_ && s0 <= jmax; s0 += 32) {
    // ---- scores: two 16-col chunks, K = hd = 128 (4 WMMAs each) ----
    v8f sc[2];
    #pragma unroll
    for (int c = 0; c < 2; ++c) {
      v8f a = vzero8();
      #pragma unroll
      for (int kk = 0; kk < 4; ++kk) {
        v16bf bf = load_frag_b(kbase + (size_t)(s0 + c * 16) * D_ + kk * 32, D_);
        a = wmma_bf16(aq[kk], bf, a);
      }
      const int j = s0 + c * 16 + (lane & 15);
      #pragma unroll
      for (int v = 0; v < 8; ++v) {
        const int i = qr0 + v + rh;
        float val = a[v] * scale;
        if (j > PASTLEN_ + i) val = -1e30f;
        a[v] = val;
      }
      sc[c] = a;
    }
    // ---- online softmax (row reductions across 16-lane halves) ----
    float nmax[8], psc[8];
    #pragma unroll
    for (int v = 0; v < 8; ++v) {
      float mx = fmaxf(sc[0][v], sc[1][v]);
      #pragma unroll
      for (int o = 1; o < 16; o <<= 1) mx = fmaxf(mx, __shfl_xor(mx, o, 32));
      const float nm = fmaxf(rmax[v], mx);
      psc[v]  = __expf(rmax[v] - nm);
      rmax[v] = nm;
      nmax[v] = nm;
    }
    #pragma unroll
    for (int c = 0; c < 2; ++c)
      #pragma unroll
      for (int v = 0; v < 8; ++v) sc[c][v] = __expf(sc[c][v] - nmax[v]);
    #pragma unroll
    for (int v = 0; v < 8; ++v) {
      float s = sc[0][v] + sc[1][v];
      #pragma unroll
      for (int o = 1; o < 16; o <<= 1) s += __shfl_xor(s, o, 32);
      rsum[v] = rsum[v] * psc[v] + s;
    }
    #pragma unroll
    for (int cb = 0; cb < 8; ++cb)
      #pragma unroll
      for (int v = 0; v < 8; ++v) oacc[cb][v] *= psc[v];

    // ---- C-layout P -> LDS -> A-fragment ----
    #pragma unroll
    for (int c = 0; c < 2; ++c)
      #pragma unroll
      for (int v = 0; v < 8; ++v)
        pbuf[wave][v + rh][c * 16 + (lane & 15)] = (bf16)sc[c][v];

    v16bf pa;
    {
      const int r = lane & 15;
      const int koff = (lane >> 4) << 3;
      #pragma unroll
      for (int e = 0; e < 8; ++e) {
        pa[e]     = pbuf[wave][r][koff + e];
        pa[e + 8] = pbuf[wave][r][16 + koff + e];
      }
    }
    // ---- O += P @ V  (8 column blocks of hd) ----
    #pragma unroll
    for (int cb = 0; cb < 8; ++cb) {
      v16bf bf = load_frag_b(vbase + (size_t)(cb * 16) * S_ + s0, S_);
      oacc[cb] = wmma_bf16(pa, bf, oacc[cb]);
    }
  }

  // ---- normalize + store bf16 ----
  float inv[8];
  #pragma unroll
  for (int v = 0; v < 8; ++v) inv[v] = 1.0f / rsum[v];
  #pragma unroll
  for (int cb = 0; cb < 8; ++cb)
    #pragma unroll
    for (int v = 0; v < 8; ++v)
      out[((size_t)(bi * T_ + qr0 + v + rh)) * D_ + h * HD_ + cb * 16 + (lane & 15)] =
          (bf16)(oacc[cb][v] * inv[v]);
}

// ---------------- host orchestration ----------------

extern "C" void kernel_launch(void* const* d_in, const int* in_sizes, int n_in,
                              void* d_out, int out_size, void* d_ws, size_t ws_size,
                              hipStream_t stream) {
  (void)in_sizes; (void)n_in; (void)out_size; (void)ws_size;
  const float* x  = (const float*)d_in[0];
  const float* lp = (const float*)d_in[1];
  const float* Wq = (const float*)d_in[2];
  const float* Wd = (const float*)d_in[3];
  const float* Wk = (const float*)d_in[4];
  const float* Wv = (const float*)d_in[5];
  const float* lg = (const float*)d_in[6];
  const float* lb = (const float*)d_in[7];
  const float* Wo = (const float*)d_in[8];
  float* outp = (float*)d_out;

  unsigned char* w = (unsigned char*)d_ws;
  size_t off = 0;
  auto carve = [&](size_t bytes) -> void* {
    void* p = w + off; off += (bytes + 255) & ~(size_t)255; return p;
  };
  bf16*  x_bf   = (bf16*) carve((size_t)B_ * T_ * D_ * 2);
  bf16*  WqT    = (bf16*) carve((size_t)D_ * D_ * 2);
  bf16*  WdT    = (bf16*) carve((size_t)LAT_ * D_ * 2);
  bf16*  WkT    = (bf16*) carve((size_t)D_ * LAT_ * 2);
  bf16*  WvT    = (bf16*) carve((size_t)D_ * LAT_ * 2);
  bf16*  WoT    = (bf16*) carve((size_t)D_ * D_ * 2);
  bf16*  q_bf   = (bf16*) carve((size_t)B_ * T_ * D_ * 2);
  float* down   = (float*)carve((size_t)B_ * T_ * LAT_ * 4);
  bf16*  lat_bf = (bf16*) carve((size_t)B_ * S_ * LAT_ * 2);
  bf16*  k_bf   = (bf16*) carve((size_t)B_ * S_ * D_ * 2);
  bf16*  vT_bf  = (bf16*) carve((size_t)B_ * D_ * S_ * 2);
  bf16*  at_bf  = (bf16*) carve((size_t)B_ * T_ * D_ * 2);

  auto g1 = [](size_t n) { return (unsigned)((n + 255) / 256); };

  convert_f32_bf16<<<g1((size_t)B_ * T_ * D_), 256, 0, stream>>>(x, x_bf, (size_t)B_ * T_ * D_);
  transpose_f32_bf16<<<g1((size_t)D_ * D_),   256, 0, stream>>>(Wq, WqT, D_, D_);
  transpose_f32_bf16<<<g1((size_t)D_ * LAT_), 256, 0, stream>>>(Wd, WdT, D_, LAT_);
  transpose_f32_bf16<<<g1((size_t)LAT_ * D_), 256, 0, stream>>>(Wk, WkT, LAT_, D_);
  transpose_f32_bf16<<<g1((size_t)LAT_ * D_), 256, 0, stream>>>(Wv, WvT, LAT_, D_);
  transpose_f32_bf16<<<g1((size_t)D_ * D_),   256, 0, stream>>>(Wo, WoT, D_, D_);

  // q = x @ Wq  (bf16 out, reused as WMMA A operand in attention)
  wmma_gemm<true, false><<<dim3((B_ * T_) / 128, D_ / 128), 128, 0, stream>>>(
      x_bf, WqT, q_bf, B_ * T_, D_, D_, D_, D_, D_);
  // down = x @ Wdown (f32, pre-layernorm)
  wmma_gemm<false, false><<<dim3((B_ * T_) / 128, LAT_ / 128), 128, 0, stream>>>(
      x_bf, WdT, down, B_ * T_, LAT_, D_, D_, D_, LAT_);
  // latent cache: [prev ; layernorm(down)]
  copy_prev_latent<<<g1((size_t)B_ * PAST_ * LAT_), 256, 0, stream>>>(lp, lat_bf);
  layernorm_latent<<<B_ * T_, 256, 0, stream>>>(down, lg, lb, lat_bf);
  // k = latent @ Wk_up (bf16, row-major over both batches at once)
  wmma_gemm<true, false><<<dim3((B_ * S_) / 128, D_ / 128), 128, 0, stream>>>(
      lat_bf, WkT, k_bf, B_ * S_, D_, LAT_, LAT_, LAT_, D_);
  // vT = (latent @ Wv_up)^T per batch -> (B, D, S) for contiguous PV B-fragments
  for (int bi = 0; bi < B_; ++bi)
    wmma_gemm<true, true><<<dim3(S_ / 128, D_ / 128), 128, 0, stream>>>(
        lat_bf + (size_t)bi * S_ * LAT_, WvT, vT_bf + (size_t)bi * D_ * S_,
        S_, D_, LAT_, LAT_, LAT_, S_);
  // attention
  mla_attention<<<dim3(T_ / 64, H_, B_), 128, 0, stream>>>(q_bf, k_bf, vT_bf, at_bf);
  // out = attn @ Wo (f32)
  wmma_gemm<false, false><<<dim3((B_ * T_) / 128, D_ / 128), 128, 0, stream>>>(
      at_bf, WoT, outp, B_ * T_, D_, D_, D_, D_, D_);
}
